// Layer_25941602468063
// MI455X (gfx1250) — matrix operand order repaired and checked by reading, not compiled, not run on hardware
//
#include <hip/hip_runtime.h>
#include <hip/hip_bf16.h>
#include <math.h>

// ---------------------------------------------------------------------------
// MI455X (gfx1250) implementation: all matmuls via v_wmma_f32_16x16x32_f16.
// B,S,DIM,HID = 256,100,512,2048.  M = B*S = 25600 tokens.
// GEMMs use a double-buffered async global->LDS stage for the weight (B)
// fragments shared by all 8 waves of a block.
// ---------------------------------------------------------------------------

typedef __attribute__((ext_vector_type(16))) _Float16 v16h;
typedef __attribute__((ext_vector_type(8)))  _Float16 v8h;
typedef __attribute__((ext_vector_type(8)))  float    v8f;
typedef __attribute__((ext_vector_type(4)))  int      v4i;

#define DIM   512
#define HID   2048
#define SVAL  100
#define BATCH 256
#define MTOK  (BATCH * SVAL)        // 25600
#define ROWS_AL (MTOK + 32)         // padded rows for f16 activation buffers
#define SPAD  112                   // 7 * 16
#define SPAD2 128                   // K-dim pad for attn @ V

// ---- async global->LDS staging (gfx1250 ASYNCcnt path), guarded -----------
#if defined(__has_builtin)
#if __has_builtin(__builtin_amdgcn_global_load_async_to_lds_b128)
#define HAVE_ASYNC_LDS 1
#endif
#endif

__device__ __forceinline__ void stage16B(const _Float16* g, _Float16* l) {
#ifdef HAVE_ASYNC_LDS
    __builtin_amdgcn_global_load_async_to_lds_b128(
        (__attribute__((address_space(1))) v4i*)g,
        (__attribute__((address_space(3))) v4i*)l, 0, 0);
#else
    *(v8h*)l = *(const v8h*)g;          // global b128 load + ds_store_b128
#endif
}

__device__ __forceinline__ void stage_wait() {
#ifdef HAVE_ASYNC_LDS
#if __has_builtin(__builtin_amdgcn_s_wait_asynccnt)
    __builtin_amdgcn_s_wait_asynccnt(0);
#else
    asm volatile("s_wait_asynccnt 0" ::: "memory");
#endif
#endif
}

__device__ __forceinline__ v8f wmma_f16(v16h a, v16h b, v8f c) {
    // D = A(16x32) * B(32x16) + C(16x16), fp32 accumulate
    return __builtin_amdgcn_wmma_f32_16x16x32_f16(
        /*neg_a=*/false, a, /*neg_b=*/false, b,
        /*c_mod=*/(short)0, c, /*reuse_a=*/false, /*reuse_b=*/false);
}

// A fragment (16x32, f16) from row-major [*, ld] f16 matrix, tile origin at A.
// Layout: lanes 0-15 -> M=lane, K=0..15; lanes 16-31 -> M=lane-16,
// K=16..31.  Two contiguous 16B loads per lane.
__device__ __forceinline__ v16h load_a_f16(const _Float16* __restrict__ A,
                                           int ld, int lane) {
    int m  = lane & 15;
    int kb = (lane >> 4) << 4;          // 0 or 16
    const _Float16* p = A + (size_t)m * ld + kb;
    v16h a;
    *(v8h*)&a       = *(const v8h*)(p);        // K = kb .. kb+7
    *((v8h*)&a + 1) = *(const v8h*)(p + 8);    // K = kb+8 .. kb+15
    return a;
}

// Same A fragment but sourced from fp32 row-major data (converted to f16).
__device__ __forceinline__ v16h load_a_f32(const float* __restrict__ A,
                                           int ld, int lane) {
    int m  = lane & 15;
    int kb = (lane >> 4) << 4;
    const float* p = A + (size_t)m * ld + kb;
    v16h a;
#pragma unroll
    for (int i = 0; i < 8; ++i) a[i]     = (_Float16)p[i];
#pragma unroll
    for (int i = 0; i < 8; ++i) a[8 + i] = (_Float16)p[8 + i];
    return a;
}

// B fragment (32x16) where B[k][n] = R[n][k] for a row-major matrix R
// (used for Q*K^T: lanes index the key row, 16 consecutive hidden values).
__device__ __forceinline__ v16h load_b_rows(const _Float16* __restrict__ R,
                                            int ld, int lane) {
    const _Float16* p = R + (size_t)(lane & 15) * ld + ((lane >> 4) << 4);
    return *(const v16h*)p;             // 32B contiguous
}

// B fragment (32x16) from row-major B[k][n] (used for attn @ V): lane = n,
// 16 strided K reads.
__device__ __forceinline__ v16h load_b_strided(const _Float16* __restrict__ Bm,
                                               int ld, int lane) {
    const _Float16* p = Bm + (size_t)((lane >> 4) << 4) * ld + (lane & 15);
    v16h b;
#pragma unroll
    for (int i = 0; i < 16; ++i) b[i] = p[(size_t)i * ld];
    return b;
}

// ---------------------------------------------------------------------------
// Repack fp32 weight W[K x N] (row-major, torch [in,out]) into f16 fragments:
// frag (nt,kt) occupies 512 contiguous halves; within a fragment the half at
// lane*16+i is W[kt*32 + (lane>>4)*16 + i][nt*16 + (lane&15)].  GEMM inner
// loops then read one contiguous v16h per lane.
// ---------------------------------------------------------------------------
__global__ void repack_w(const float* __restrict__ W, _Float16* __restrict__ P,
                         int K, int N) {
    int total = K * N;
    int kt32  = K >> 5;
    for (int p = blockIdx.x * blockDim.x + threadIdx.x; p < total;
         p += gridDim.x * blockDim.x) {
        int frag = p >> 9;
        int r    = p & 511;
        int lane = r >> 4, i = r & 15;
        int nt = frag / kt32, kt = frag % kt32;
        int n = (nt << 4) + (lane & 15);
        int k = (kt << 5) + ((lane >> 4) << 4) + i;
        P[p] = (_Float16)W[(size_t)k * N + n];
    }
}

// Zero the pad rows (tokens 25600..25631) of the f16 activation buffers so
// out-of-range attention tile reads are finite.
__global__ void init_pads(_Float16* h, _Float16* Qb, _Float16* Kb, _Float16* Vb) {
    int n = 32 * HID;
    size_t base = (size_t)MTOK * HID;
    for (int i = blockIdx.x * blockDim.x + threadIdx.x; i < n;
         i += gridDim.x * blockDim.x) {
        h[base + i]  = (_Float16)0.f;
        Qb[base + i] = (_Float16)0.f;
        Kb[base + i] = (_Float16)0.f;
        Vb[base + i] = (_Float16)0.f;
    }
}

// ---------------------------------------------------------------------------
// Generic WMMA GEMM:  Out[M x N] = act(A[M x K] * Wpack + bias (+ resid))
// Block = 256 threads (8 waves); block tile = 128(M) x 64(N), each wave owns a
// 16 x 64 strip.  The 64-column B slab for one K-step (4 frags x 512 halves =
// 4 KB) is staged into LDS once per block, double-buffered: the async copy of
// K-step kt+1 overlaps the WMMAs of kt.  B fragments then come from ds_load.
// ---------------------------------------------------------------------------
template <bool AF32, bool RELU, bool RESID>
__global__ void __launch_bounds__(256)
gemm_wmma(const void* __restrict__ Ap, const _Float16* __restrict__ Bp,
          const float* __restrict__ bias, const float* __restrict__ resid,
          void* __restrict__ Outp, int K, int N) {
    __shared__ _Float16 Bs[2][4 * 512];          // 2 x 4KB double buffer

    const int lane = threadIdx.x & 31;
    const int wave = threadIdx.x >> 5;
    const int row0 = blockIdx.x * 128 + wave * 16;
    const int col0 = blockIdx.y * 64;
    const int kt32 = K >> 5;
    const int nf0  = col0 >> 4;

    // staging role of this thread: frag j (0..3), 16B element e (0..63)
    const int sj = threadIdx.x >> 6;
    const int se = (threadIdx.x & 63) * 8;       // half offset within frag

    // prologue: stage kt = 0 into buffer 0
    stage16B(Bp + ((size_t)(nf0 + sj) * kt32 + 0) * 512 + se, &Bs[0][sj * 512 + se]);
    stage_wait();
    __syncthreads();

    v8f acc[4] = {};
    for (int kt = 0; kt < kt32; ++kt) {
        const int cur = kt & 1, nxt = cur ^ 1;
        // kick off async stage of the next K-slab while computing this one
        if (kt + 1 < kt32)
            stage16B(Bp + ((size_t)(nf0 + sj) * kt32 + (kt + 1)) * 512 + se,
                     &Bs[nxt][sj * 512 + se]);

        v16h a;
        if (AF32) {
            a = load_a_f32((const float*)Ap + (size_t)row0 * K + kt * 32, K, lane);
        } else {
            a = load_a_f16((const _Float16*)Ap + (size_t)row0 * K + kt * 32, K, lane);
            if (kt + 8 < kt32) {
                const _Float16* pf = (const _Float16*)Ap +
                    (size_t)(row0 + (lane & 15)) * K + (kt + 8) * 32;
                __builtin_prefetch(pf, 0, 1);    // global_prefetch_b8
            }
        }
#pragma unroll
        for (int j = 0; j < 4; ++j) {
            v16h b = *(const v16h*)(&Bs[cur][j * 512 + lane * 16]);  // ds_load
            acc[j] = wmma_f16(a, b, acc[j]);
        }
        stage_wait();        // this wave's async copies into nxt complete
        __syncthreads();     // all waves done reading cur + done staging nxt
    }

    // C/D layout: lane L, reg r -> M = r + (L>=16 ? 8 : 0), N = L & 15
    const int mrow = row0 + ((lane >> 4) << 3);
    const int ncol = col0 + (lane & 15);
#pragma unroll
    for (int j = 0; j < 4; ++j) {
#pragma unroll
        for (int r = 0; r < 8; ++r) {
            int m = mrow + r, n = ncol + j * 16;
            float v = acc[j][r] + bias[n];
            if (RELU) v = v > 0.f ? v : 0.f;
            if (RESID)
                ((float*)Outp)[(size_t)m * N + n] = v + resid[(size_t)m * N + n];
            else
                ((_Float16*)Outp)[(size_t)m * N + n] = (_Float16)v;
        }
    }
}

// ---------------------------------------------------------------------------
// Attention: one workgroup (8 waves) per batch element.
//   phase 1: scores = Q K^T / sqrt(512) via WMMA (7x7 tiles of 16x16)
//   phase 2: masked softmax (mask: key j <= i+1), scores kept f16 in LDS
//   phase 3: O = attn @ V via WMMA; attn A-fragments served from LDS
// ---------------------------------------------------------------------------
__global__ void __launch_bounds__(256)
attn_kernel(const _Float16* __restrict__ Q, const _Float16* __restrict__ Km,
            const _Float16* __restrict__ V, _Float16* __restrict__ O) {
    __shared__ _Float16 attnS[SPAD][SPAD2];   // 112 x 128 f16 = 28 KB LDS

    const int b    = blockIdx.x;
    const int lane = threadIdx.x & 31;
    const int wave = threadIdx.x >> 5;
    const size_t base = (size_t)b * SVAL * HID;
    const float scale = 0.04419417382415922f;   // 1/sqrt(512)

    // ---- phase 1: logits ----
    for (int t = wave; t < 49; t += 8) {
        int ti = t / 7, tj = t % 7;
        v8f acc = {};
        const _Float16* qp = Q  + base + (size_t)(ti * 16) * HID;
        const _Float16* kp = Km + base + (size_t)(tj * 16) * HID;
        for (int kt = 0; kt < HID / 32; ++kt) {
            v16h a  = load_a_f16(qp + kt * 32, HID, lane);
            v16h bf = load_b_rows(kp + kt * 32, HID, lane);
            acc = wmma_f16(a, bf, acc);
        }
        int q = ti * 16 + ((lane >> 4) << 3);
        int k = tj * 16 + (lane & 15);
#pragma unroll
        for (int r = 0; r < 8; ++r)
            attnS[q + r][k] = (_Float16)(acc[r] * scale);
    }
    __syncthreads();

    // ---- phase 2: masked softmax, one thread per row ----
    for (int r = threadIdx.x; r < SPAD; r += 256) {
        if (r < SVAL) {
            int vl = (r + 2 < SVAL) ? r + 2 : SVAL;   // tril(k=1)
            float mx = -1e30f;
            for (int k = 0; k < vl; ++k) mx = fmaxf(mx, (float)attnS[r][k]);
            float s = 0.f;
            for (int k = 0; k < vl; ++k) s += __expf((float)attnS[r][k] - mx);
            float inv = 1.f / s;
            for (int k = 0; k < vl; ++k)
                attnS[r][k] = (_Float16)(__expf((float)attnS[r][k] - mx) * inv);
            for (int k = vl; k < SPAD2; ++k) attnS[r][k] = (_Float16)0.f;
        } else {
            for (int k = 0; k < SPAD2; ++k) attnS[r][k] = (_Float16)0.f;
        }
    }
    __syncthreads();

    // ---- phase 3: O = attn @ V ----
    for (int nf = wave; nf < HID / 16; nf += 8) {
        v16h bf[4];
#pragma unroll
        for (int kt = 0; kt < 4; ++kt)
            bf[kt] = load_b_strided(V + base + (size_t)(kt * 32) * HID + nf * 16,
                                    HID, lane);
        for (int mt = 0; mt < 7; ++mt) {
            v8f acc = {};
#pragma unroll
            for (int kt = 0; kt < 4; ++kt) {
                v16h a = load_a_f16(&attnS[mt * 16][kt * 32], SPAD2, lane);
                acc = wmma_f16(a, bf[kt], acc);
            }
            int q = mt * 16 + ((lane >> 4) << 3);
            int n = nf * 16 + (lane & 15);
#pragma unroll
            for (int r = 0; r < 8; ++r) {
                int qq = q + r;
                if (qq < SVAL)
                    O[base + (size_t)qq * HID + n] = (_Float16)acc[r];
            }
        }
    }
}

// ---------------------------------------------------------------------------
// LayerNorm over the full (S, DIM) = 51200-element slab per batch element.
// ---------------------------------------------------------------------------
__global__ void __launch_bounds__(256)
reduce_stats(const float* __restrict__ y, float* __restrict__ stats) {
    __shared__ float ssum[256], ssq[256];
    const int b = blockIdx.x;
    const float* p = y + (size_t)b * (SVAL * DIM);
    float s = 0.f, q = 0.f;
    for (int i = threadIdx.x; i < SVAL * DIM; i += 256) {
        float v = p[i];
        s += v; q += v * v;
    }
    ssum[threadIdx.x] = s; ssq[threadIdx.x] = q;
    __syncthreads();
    for (int st = 128; st > 0; st >>= 1) {
        if (threadIdx.x < st) {
            ssum[threadIdx.x] += ssum[threadIdx.x + st];
            ssq[threadIdx.x]  += ssq[threadIdx.x + st];
        }
        __syncthreads();
    }
    if (threadIdx.x == 0) {
        float n = (float)(SVAL * DIM);
        float mean = ssum[0] / n;
        float var  = ssq[0] / n - mean * mean;
        stats[b * 2]     = mean;
        stats[b * 2 + 1] = var;
    }
}

__global__ void __launch_bounds__(256)
norm_kernel(const float* __restrict__ y, const float* __restrict__ stats,
            float* __restrict__ out) {
    const int b = blockIdx.y;
    const int i = blockIdx.x * 256 + threadIdx.x;     // < 51200
    float mean = stats[b * 2], var = stats[b * 2 + 1];
    float inv = rsqrtf(var + 1e-5f);
    size_t idx = (size_t)b * (SVAL * DIM) + i;
    out[idx] = (y[idx] - mean) * inv;
}

// ---------------------------------------------------------------------------
// Host-side launch
// ---------------------------------------------------------------------------
extern "C" void kernel_launch(void* const* d_in, const int* in_sizes, int n_in,
                              void* d_out, int out_size, void* d_ws, size_t ws_size,
                              hipStream_t stream) {
    (void)in_sizes; (void)n_in; (void)out_size; (void)ws_size;

    const float* x    = (const float*)d_in[0];
    const float* W_w  = (const float*)d_in[1];
    const float* W_b  = (const float*)d_in[2];
    const float* WK_w = (const float*)d_in[3];
    const float* WK_b = (const float*)d_in[4];
    const float* WQ_w = (const float*)d_in[5];
    const float* WQ_b = (const float*)d_in[6];
    const float* WV_w = (const float*)d_in[7];
    const float* WV_b = (const float*)d_in[8];
    const float* W2_w = (const float*)d_in[9];
    const float* W2_b = (const float*)d_in[10];
    float* out = (float*)d_out;

    // workspace carve-out (256B aligned slabs)
    size_t off = 0;
    auto carve = [&](size_t bytes) {
        void* p = (char*)d_ws + off;
        off += (bytes + 255) & ~(size_t)255;
        return p;
    };
    _Float16* Wp  = (_Float16*)carve((size_t)DIM * HID * 2);
    _Float16* WKp = (_Float16*)carve((size_t)HID * HID * 2);
    _Float16* WQp = (_Float16*)carve((size_t)HID * HID * 2);
    _Float16* WVp = (_Float16*)carve((size_t)HID * HID * 2);
    _Float16* W2p = (_Float16*)carve((size_t)HID * DIM * 2);
    _Float16* h   = (_Float16*)carve((size_t)ROWS_AL * HID * 2);  // reused as O
    _Float16* Qb  = (_Float16*)carve((size_t)ROWS_AL * HID * 2);
    _Float16* Kb  = (_Float16*)carve((size_t)ROWS_AL * HID * 2);
    _Float16* Vb  = (_Float16*)carve((size_t)ROWS_AL * HID * 2);
    float*    y   = (float*)carve((size_t)MTOK * DIM * 4);
    float*  stats = (float*)carve((size_t)BATCH * 2 * 4);

    // 1) weight repack (fp32 -> f16 B-fragment layout)
    repack_w<<<2048, 256, 0, stream>>>(W_w,  Wp,  DIM, HID);
    repack_w<<<4096, 256, 0, stream>>>(WK_w, WKp, HID, HID);
    repack_w<<<4096, 256, 0, stream>>>(WQ_w, WQp, HID, HID);
    repack_w<<<4096, 256, 0, stream>>>(WV_w, WVp, HID, HID);
    repack_w<<<2048, 256, 0, stream>>>(W2_w, W2p, HID, DIM);
    init_pads<<<256, 256, 0, stream>>>(h, Qb, Kb, Vb);

    // 2) h = relu(x @ W + b)          M=25600, K=512,  N=2048
    gemm_wmma<true, true, false>
        <<<dim3(MTOK / 128, HID / 64), 256, 0, stream>>>(x, Wp, W_b, nullptr, h,
                                                         DIM, HID);
    // 3) K/Q/V = h @ W* + b*          M=25600, K=2048, N=2048
    gemm_wmma<false, false, false>
        <<<dim3(MTOK / 128, HID / 64), 256, 0, stream>>>(h, WKp, WK_b, nullptr, Kb,
                                                         HID, HID);
    gemm_wmma<false, false, false>
        <<<dim3(MTOK / 128, HID / 64), 256, 0, stream>>>(h, WQp, WQ_b, nullptr, Qb,
                                                         HID, HID);
    gemm_wmma<false, false, false>
        <<<dim3(MTOK / 128, HID / 64), 256, 0, stream>>>(h, WVp, WV_b, nullptr, Vb,
                                                         HID, HID);

    // 4) attention (O aliases h; h is dead after K/Q/V)
    _Float16* O = h;
    attn_kernel<<<BATCH, 256, 0, stream>>>(Qb, Kb, Vb, O);

    // 5) y = O @ W2 + b2 + x          M=25600, K=2048, N=512 (fp32 out)
    gemm_wmma<false, false, true>
        <<<dim3(MTOK / 128, DIM / 64), 256, 0, stream>>>(O, W2p, W2_b, x, y,
                                                         HID, DIM);

    // 6) LayerNorm over (S, DIM) per batch element
    reduce_stats<<<BATCH, 256, 0, stream>>>(y, stats);
    norm_kernel<<<dim3((SVAL * DIM) / 256, BATCH), 256, 0, stream>>>(y, stats, out);
}
